// PartDeformEncoder2_25555055411687
// MI455X (gfx1250) — matrix operand order repaired and checked by reading, not compiled, not run on hardware
//
#include <hip/hip_runtime.h>
#include <hip/hip_bf16.h>

// ---------------------------------------------------------------------------
// PartDeformEncoder2 for MI455X (gfx1250, wave32, WMMA)
//
//   deg/rnorm -> 3x fused GCN layers (xw+selfloop, edge scatter-add atomics,
//   rnorm*acc+bias+tanh) -> two split-K bf16 WMMA GEMMs into a shared f32
//   accumulator (concat via column offset) -> tanh+bias -> final bf16 WMMA
//   GEMM into d_out via atomic f32 adds.
//
// Roofline: GEMM weight streams (151 MB fp32, read once) dominate ->
// ~6.5us floor at 23.3 TB/s. v_wmma_f32_16x16x32_bf16 keeps math off the
// critical path; f32->bf16 uses v_cvt_pk_bf16_f32 (packed u32 LDS stores),
// double-buffered LDS = 1 barrier / K-step.
// ---------------------------------------------------------------------------

#define B_  128
#define N_  8192
#define E_  131072
#define FEAT_ 512
#define BN_ (B_ * N_)

typedef __attribute__((ext_vector_type(16))) __bf16 v16bf;
typedef __attribute__((ext_vector_type(2)))  __bf16 v2bf;
typedef __attribute__((ext_vector_type(8)))  float  v8f;

union F4 { float4 v; float a[4]; };

// pack two floats into two bf16 halves of a u32
#if __has_builtin(__builtin_amdgcn_cvt_pk_bf16_f32)
__device__ __forceinline__ unsigned pkbf(float a, float b) {
  union { v2bf v; unsigned u; } o;
  o.v = __builtin_amdgcn_cvt_pk_bf16_f32(a, b);   // single VALU op
  return o.u;
}
#else
__device__ __forceinline__ unsigned pkbf(float a, float b) {
  union { struct { __bf16 l, h; } p; unsigned u; } o;
  o.p.l = (__bf16)a;
  o.p.h = (__bf16)b;
  return o.u;
}
#endif

// ---------------------------------------------------------------------------
// Split-K bf16 WMMA GEMM: C[128 x 64chunk] += A[128 x K] * W[K x ldw]
// grid.x = column chunk (64 cols), grid.y = K split (kChunk each), 256 thr.
// LDS tiles are stored pre-swizzled into WMMA fragment order (ISA 7.12.2):
//   A (16x32 bf16): lane = (m&15) | (k[3]<<4), elem = (k&7) | (k[4]<<3)
//   B (32x16 bf16): lane = (n&15) | (k[3]<<4), elem = (k&7) | (k[4]<<3)
// so each lane's 16-element fragment is 32 contiguous bytes (2x ds_load_b128).
// ---------------------------------------------------------------------------
__global__ __launch_bounds__(256) void k_gemm_bf16(
    const float* __restrict__ A, int lda,
    const float* __restrict__ W, int ldw,
    float* __restrict__ accOut, int ldacc, int colOff, int kChunk)
{
  __shared__ __align__(32) __bf16 Abuf[2][8 * 512];  // 8 M-tiles, double buffer
  __shared__ __align__(32) __bf16 Bbuf[2][4 * 512];  // 4 N-tiles, double buffer

  const int tid   = threadIdx.x;
  const int lane  = tid & 31;
  const int wv    = tid >> 5;                        // wave id -> M-tile
  const int ncol0 = blockIdx.x * 64;
  const int kIters = kChunk >> 5;
  const int kBase0 = blockIdx.y * kChunk;

  // A staging: 16 elems/thread -> row am, k in [ak, ak+16)
  const int am = tid >> 1;
  const int ak = (tid & 1) << 4;
  // B staging: 8 elems/thread -> two adjacent k rows (2*kp, 2*kp+1), 4 cols
  const int kp  = tid >> 4;                          // 0..15
  const int nc4 = (tid & 15) << 2;                   // 0..60

  const int aTile = (am >> 4) << 8;                  // u32-index tile base

  v8f acc0 = {}, acc1 = {}, acc2 = {}, acc3 = {};

  auto stageA = [&](int buf, int kb) {
    const float* ap = A + (size_t)am * lda + kb + ak;
    unsigned* dst = reinterpret_cast<unsigned*>(Abuf[buf]);
    #pragma unroll
    for (int j = 0; j < 16; j += 4) {
      F4 f; f.v = *reinterpret_cast<const float4*>(ap + j);
      const int k0   = ak + j;                               // even, group of 4
      const int la   = (am & 15) | (((k0 >> 3) & 1) << 4);
      const int ep0  = ((k0 & 7) | ((k0 & 16) >> 1)) >> 1;   // u32 slot
      dst[aTile | (la << 3) | ep0]       = pkbf(f.a[0], f.a[1]);
      dst[aTile | (la << 3) | (ep0 + 1)] = pkbf(f.a[2], f.a[3]);
    }
  };
  auto stageB = [&](int buf, int kb) {
    const int k = kp << 1;
    const float* w0 = W + (size_t)(kb + k) * ldw + ncol0 + nc4;
    F4 r0, r1;
    r0.v = *reinterpret_cast<const float4*>(w0);
    r1.v = *reinterpret_cast<const float4*>(w0 + ldw);
    unsigned* dst = reinterpret_cast<unsigned*>(Bbuf[buf]);
    const int ep = ((k & 7) | ((k & 16) >> 1)) >> 1;
    const int lb = ((k >> 3) & 1) << 4;
    #pragma unroll
    for (int j = 0; j < 4; ++j) {
      const int nc = nc4 + j;
      dst[((nc >> 4) << 8) | ((((nc & 15) | lb)) << 3) | ep] = pkbf(r0.a[j], r1.a[j]);
    }
  };

  stageA(0, kBase0);
  stageB(0, kBase0);
  __syncthreads();

  for (int it = 0; it < kIters; ++it) {
    const int cur = it & 1;

    // fragment ds_loads first: they only depend on the previous barrier
    const v16bf a  = *reinterpret_cast<const v16bf*>(&Abuf[cur][(wv << 9) | (lane << 4)]);
    const v16bf b0 = *reinterpret_cast<const v16bf*>(&Bbuf[cur][(0 << 9) | (lane << 4)]);
    const v16bf b1 = *reinterpret_cast<const v16bf*>(&Bbuf[cur][(1 << 9) | (lane << 4)]);
    const v16bf b2 = *reinterpret_cast<const v16bf*>(&Bbuf[cur][(2 << 9) | (lane << 4)]);
    const v16bf b3 = *reinterpret_cast<const v16bf*>(&Bbuf[cur][(3 << 9) | (lane << 4)]);

    // stage next tile into the other buffer (global loads + packed ds stores)
    if (it + 1 < kIters) {
      const int kn = kBase0 + ((it + 1) << 5);
      stageA(cur ^ 1, kn);
      stageB(cur ^ 1, kn);
      if (it + 2 < kIters)  // warm L2 for the tile after next
        __builtin_prefetch(W + (size_t)(kBase0 + ((it + 2) << 5) + (kp << 1)) * ldw + ncol0 + nc4, 0, 1);
    }

    acc0 = __builtin_amdgcn_wmma_f32_16x16x32_bf16(false, a, false, b0, (short)0, acc0, false, false);
    acc1 = __builtin_amdgcn_wmma_f32_16x16x32_bf16(false, a, false, b1, (short)0, acc1, false, false);
    acc2 = __builtin_amdgcn_wmma_f32_16x16x32_bf16(false, a, false, b2, (short)0, acc2, false, false);
    acc3 = __builtin_amdgcn_wmma_f32_16x16x32_bf16(false, a, false, b3, (short)0, acc3, false, false);
    __syncthreads();
  }

  // Epilogue: C/D layout (ISA 7.12.2) -> VGPR j = row j (lanes 0-15) / j+8
  const int row0 = (wv << 4) + ((lane >> 4) << 3);
  const int col  = colOff + ncol0 + (lane & 15);
  #pragma unroll
  for (int j = 0; j < 8; ++j) {
    float* base = accOut + (size_t)(row0 + j) * ldacc + col;
    atomicAdd(base + 0,  acc0[j]);
    atomicAdd(base + 16, acc1[j]);
    atomicAdd(base + 32, acc2[j]);
    atomicAdd(base + 48, acc3[j]);
  }
}

// ---------------------------------------------------------------------------
// GCN helper kernels
// ---------------------------------------------------------------------------
__global__ void k_fill(float* __restrict__ p, int n, float v) {
  int i = blockIdx.x * 256 + threadIdx.x;
  if (i < n) p[i] = v;
}

__global__ void k_deg(const long long* __restrict__ dst, float* __restrict__ deg, int E) {
  int e = blockIdx.x * 256 + threadIdx.x;
  if (e < E) atomicAdd(&deg[(int)dst[e]], 1.0f);
}

__global__ void k_rnorm(float* __restrict__ d, int n) {
  int i = blockIdx.x * 256 + threadIdx.x;
  if (i < n) d[i] = rsqrtf(d[i]);
}

// y = (x*scale) @ W  scaled by rnorm[src]; also seeds ACC with self-loop term.
template <int CIN, int COUT>
__global__ void k_xw(const float* __restrict__ x, int xStride, float scale,
                     const float* __restrict__ W, const float* __restrict__ rnorm,
                     float* __restrict__ Y, float* __restrict__ ACC)
{
  int i = blockIdx.x * 256 + threadIdx.x;   // i = b*N + n  (exact grid)
  int n = i & (N_ - 1);
  float r = rnorm[n];
  const float* xp = x + (size_t)i * xStride;
  float xv[CIN];
  #pragma unroll
  for (int c = 0; c < CIN; ++c) xv[c] = xp[c] * scale;
  #pragma unroll
  for (int co = 0; co < COUT; ++co) {
    float s = 0.f;
    #pragma unroll
    for (int ci = 0; ci < CIN; ++ci) s += xv[ci] * W[ci * COUT + co];
    s *= r;
    Y[(size_t)i * COUT + co]   = s;
    ACC[(size_t)i * COUT + co] = s;   // self-loop contribution
  }
}

template <int C>
__global__ void k_scatter(const long long* __restrict__ src,
                          const long long* __restrict__ dst,
                          const float* __restrict__ Y, float* __restrict__ ACC, int E)
{
  int e = blockIdx.x * 256 + threadIdx.x;
  int b = blockIdx.y;
  if (e >= E) return;
  int s = (int)src[e], d = (int)dst[e];
  const float* yp = Y + ((size_t)b * N_ + s) * C;
  float*       ap = ACC + ((size_t)b * N_ + d) * C;
  #pragma unroll
  for (int c = 0; c < C; ++c) atomicAdd(&ap[c], yp[c]);
}

template <int C>
__global__ void k_finish(const float* __restrict__ ACC, const float* __restrict__ rnorm,
                         const float* __restrict__ bias, float* __restrict__ out)
{
  int i = blockIdx.x * 256 + threadIdx.x;
  int n = i & (N_ - 1);
  float r = rnorm[n];
  #pragma unroll
  for (int c = 0; c < C; ++c)
    out[(size_t)i * C + c] = tanhf(r * ACC[(size_t)i * C + c] + bias[c]);
}

// net = tanh(netacc + concat(bm_logr, bm_s))
__global__ void k_net_finalize(const float* __restrict__ acc,
                               const float* __restrict__ b_logr,
                               const float* __restrict__ b_s,
                               float* __restrict__ net)
{
  int i = blockIdx.x * 256 + threadIdx.x;     // 0 .. 128*1024-1
  int c = i & 1023;
  float bias = (c < FEAT_) ? b_logr[c] : b_s[c - FEAT_];
  net[i] = tanhf(acc[i] + bias);
}

// ---------------------------------------------------------------------------
// Host-side orchestration
// ---------------------------------------------------------------------------
extern "C" void kernel_launch(void* const* d_in, const int* in_sizes, int n_in,
                              void* d_out, int out_size, void* d_ws, size_t ws_size,
                              hipStream_t stream) {
  const float* featurein = (const float*)d_in[0];
  const float* W1_logr   = (const float*)d_in[1];
  const float* b1_logr   = (const float*)d_in[2];
  const float* W1_s      = (const float*)d_in[3];
  const float* b1_s      = (const float*)d_in[4];
  const float* W4_s      = (const float*)d_in[5];
  const float* b4_s      = (const float*)d_in[6];
  const float* Wm_logr   = (const float*)d_in[7];
  const float* bm_logr   = (const float*)d_in[8];
  const float* Wm_s      = (const float*)d_in[9];
  const float* bm_s      = (const float*)d_in[10];
  const float* Wmu       = (const float*)d_in[11];
  const long long* eidx  = (const long long*)d_in[12];
  const long long* esrc  = eidx;
  const long long* edst  = eidx + E_;
  float* out = (float*)d_out;

  // workspace layout (floats)
  float* ws = (float*)d_ws;
  size_t off = 0;
  float* rnorm  = ws + off; off += N_;                 // 8192
  float* Y      = ws + off; off += (size_t)BN_ * 6;    // xw staging (per layer)
  float* ACC    = ws + off; off += (size_t)BN_ * 6;    // scatter accumulator
  float* A1     = ws + off; off += (size_t)BN_ * 3;    // net_logr  [B, N*3]
  float* T      = ws + off; off += (size_t)BN_ * 6;    // net_s     [B, N*6]
  float* netacc = ws + off; off += (size_t)B_ * 1024;  // GEMM1|2 accumulator
  float* net    = ws + off; off += (size_t)B_ * 1024;  // tanh'd concat
  if (ws_size < off * sizeof(float)) return;           // insufficient scratch

  const int BLK = 256;
  const dim3 gBN(BN_ / BLK), gE(E_ / BLK, B_);

  // degrees -> rnorm (deg starts at 1.0 for self-loop)
  k_fill<<<N_ / BLK, BLK, 0, stream>>>(rnorm, N_, 1.0f);
  k_deg<<<E_ / BLK, BLK, 0, stream>>>(edst, rnorm, E_);
  k_rnorm<<<N_ / BLK, BLK, 0, stream>>>(rnorm, N_);

  // zero GEMM accumulators (graph-replay safe: re-zeroed every call)
  k_fill<<<(B_ * 1024) / BLK, BLK, 0, stream>>>(netacc, B_ * 1024, 0.0f);
  k_fill<<<(B_ * FEAT_) / BLK, BLK, 0, stream>>>(out, B_ * FEAT_, 0.0f);

  // --- GCN layer: logr (Cin=3, Cout=3, x = featurein[...,0:3]/4) ---
  k_xw<3, 3><<<gBN, BLK, 0, stream>>>(featurein + 0, 9, 0.25f, W1_logr, rnorm, Y, ACC);
  k_scatter<3><<<gE, BLK, 0, stream>>>(esrc, edst, Y, ACC, E_);
  k_finish<3><<<gBN, BLK, 0, stream>>>(ACC, rnorm, b1_logr, A1);

  // --- GCN layer: s #1 (Cin=6, Cout=6, x = featurein[...,3:9]/50) ---
  k_xw<6, 6><<<gBN, BLK, 0, stream>>>(featurein + 3, 9, 0.02f, W1_s, rnorm, Y, ACC);
  k_scatter<6><<<gE, BLK, 0, stream>>>(esrc, edst, Y, ACC, E_);
  k_finish<6><<<gBN, BLK, 0, stream>>>(ACC, rnorm, b1_s, T);

  // --- GCN layer: s #2 (input T, output overwrites T after consumption) ---
  k_xw<6, 6><<<gBN, BLK, 0, stream>>>(T, 6, 1.0f, W4_s, rnorm, Y, ACC);
  k_scatter<6><<<gE, BLK, 0, stream>>>(esrc, edst, Y, ACC, E_);
  k_finish<6><<<gBN, BLK, 0, stream>>>(ACC, rnorm, b4_s, T);

  // --- big GEMMs (bf16 WMMA, split-K, atomic f32 merge) ---
  // C1 = A1[128,24576] @ Wm_logr[24576,512] -> netacc cols [0,512)
  k_gemm_bf16<<<dim3(FEAT_ / 64, 24), BLK, 0, stream>>>(A1, N_ * 3, Wm_logr, FEAT_,
                                                        netacc, 1024, 0, 1024);
  // C2 = T[128,49152] @ Wm_s[49152,512] -> netacc cols [512,1024)
  k_gemm_bf16<<<dim3(FEAT_ / 64, 48), BLK, 0, stream>>>(T, N_ * 6, Wm_s, FEAT_,
                                                        netacc, 1024, FEAT_, 1024);
  // net = tanh(netacc + bias)
  k_net_finalize<<<(B_ * 1024) / BLK, BLK, 0, stream>>>(netacc, bm_logr, bm_s, net);

  // mu = net[128,1024] @ Wmu[1024,512] -> d_out (zeroed above)
  k_gemm_bf16<<<dim3(FEAT_ / 64, 1), BLK, 0, stream>>>(net, 1024, Wmu, FEAT_,
                                                       out, FEAT_, 0, 1024);
}
